// Net_35304631173765
// MI455X (gfx1250) — matrix operand order repaired and checked by reading, not compiled, not run on hardware
//
#include <hip/hip_runtime.h>

// ---------------------------------------------------------------------------
// MI455X (gfx1250) implementation.
//  - local conv  -> per-bucket GEMM, v_wmma_f32_16x16x32_f16
//                   A tile DMA'd into LDS by the Tensor Data Mover (overlapped
//                   with the im2col B gather), tracked with s_wait_tensorcnt
//  - 1x1 convs   -> WMMA GEMM over pixels (concat is free via buffer layout)
//  - 3x3 convs   -> scalar VALU (negligible FLOPs)
// ---------------------------------------------------------------------------

typedef __attribute__((ext_vector_type(16))) _Float16 v16h;
typedef __attribute__((ext_vector_type(8)))  float    v8f;

#define NFX   64
#define NTB   72
#define HWD   48
#define NPIX  2304            // 48*48
#define KW    576             // 64*3*3 local-conv weight taps
#define KLOC  577             // + bias column
#define EMBD  (64 * 577)      // per-bucket embedding row

#if __has_builtin(__builtin_amdgcn_tensor_load_to_lds) && \
    __has_builtin(__builtin_amdgcn_s_wait_tensorcnt)
#define HAVE_TDM 1
typedef unsigned u32x4 __attribute__((ext_vector_type(4)));
typedef int      i32x8 __attribute__((ext_vector_type(8)));
typedef int      i32x4 __attribute__((ext_vector_type(4)));
#else
#define HAVE_TDM 0
#endif

__device__ __forceinline__ int reflect48(int t) {
  t = (t < 0) ? -t : t;
  return (t > 47) ? (94 - t) : t;
}

// K-index maps for wave32 v_wmma_f32_16x16x32_f16 fragments (ISA 7.12.2):
// A (16x32, 16b): lanes 0-15 rows M, K pairs 0..7 in v0-3, 16..23 in v4-7;
//                 lanes 16-31 same rows, K+8 / K+24.
__device__ __forceinline__ int a_kidx(int hi, int e) {
  int v = e >> 1, h = e & 1;
  return ((v & 4) ? 16 : 0) + (hi ? 8 : 0) + ((v & 3) << 1) + h;
}
// B (32x16, 16b): same K order as A: lanes 0-15 K 0..7,16..23; lanes 16-31 K+8.
__device__ __forceinline__ int b_kidx(int hi, int e) {
  int v = e >> 1, h = e & 1;
  return ((v & 4) ? 16 : 0) + (hi ? 8 : 0) + ((v & 3) << 1) + h;
}

union AFrag { v16h v; _Float16 h[16]; };
union CFrag { v8f  v; float    f[8];  };

// ---------------------------------------------------------------------------
// bucket binning (pixlist ordering is nondeterministic; outputs are not)
// ---------------------------------------------------------------------------
__global__ void k_zero_bins(int* counts, int* cursor, int* offsets) {
  int t = threadIdx.x;
  if (t < NTB) { counts[t] = 0; cursor[t] = 0; }
  if (t <= NTB) offsets[t] = 0;
}

__global__ void k_hist(const int* __restrict__ buckets, int* counts) {
  int p = blockIdx.x * blockDim.x + threadIdx.x;
  if (p < NPIX) atomicAdd(&counts[buckets[p]], 1);
}

__global__ void k_scan(const int* __restrict__ counts, int* offsets) {
  if (blockIdx.x == 0 && threadIdx.x == 0) {
    int s = 0;
    for (int i = 0; i < NTB; ++i) { offsets[i] = s; s += counts[i]; }
    offsets[NTB] = s;
  }
}

__global__ void k_fill(const int* __restrict__ buckets, const int* __restrict__ offsets,
                       int* cursor, int* pixlist) {
  int p = blockIdx.x * blockDim.x + threadIdx.x;
  if (p < NPIX) {
    int b = buckets[p];
    int pos = atomicAdd(&cursor[b], 1);
    pixlist[offsets[b] + pos] = p;
  }
}

__global__ void k_copy(float* __restrict__ dst, const float* __restrict__ src, int n) {
  int i = blockIdx.x * blockDim.x + threadIdx.x;
  if (i < n) dst[i] = src[i];
}

// ---------------------------------------------------------------------------
// scalar 3x3 conv (zero pad), generic groups; entry / exit / grouped body1
// ---------------------------------------------------------------------------
__global__ void k_conv3x3(const float* __restrict__ in, const float* __restrict__ w,
                          const float* __restrict__ bias, float* __restrict__ out,
                          int Cin, int Cout, int groups, int do_relu) {
  int idx = blockIdx.x * blockDim.x + threadIdx.x;
  if (idx >= Cout * NPIX) return;
  int o = idx / NPIX, p = idx - o * NPIX;
  int y = p / HWD, x = p - y * HWD;
  int cig = Cin / groups;
  int g = o / (Cout / groups);
  float s = bias[o];
  for (int ci = 0; ci < cig; ++ci) {
    const float* inc = in + (g * cig + ci) * NPIX;
    const float* wc  = w + (o * cig + ci) * 9;
#pragma unroll
    for (int i = 0; i < 3; ++i) {
      int yy = y + i - 1;
      if (yy < 0 || yy >= HWD) continue;
#pragma unroll
      for (int j = 0; j < 3; ++j) {
        int xx = x + j - 1;
        if (xx < 0 || xx >= HWD) continue;
        s += wc[i * 3 + j] * inc[yy * HWD + xx];
      }
    }
  }
  if (do_relu) s = fmaxf(s, 0.f);
  out[o * NPIX + p] = s;
}

// ---------------------------------------------------------------------------
// 1x1 conv as WMMA GEMM: out[64, NPIX] = relu(W[64,Cin] x in[Cin,NPIX] + b (+res))
// grid = (4 m-tiles, 144 n-tiles), 32 threads (one wave) per block.
// Templated on HAS_RES -> branchless epilogue.
// ---------------------------------------------------------------------------
template <bool HAS_RES>
__global__ void k_conv1x1_wmma(const float* __restrict__ W, const float* __restrict__ bias,
                               const float* __restrict__ in, const float* __restrict__ res,
                               float* __restrict__ out, int Cin) {
  int lane = threadIdx.x;
  int hi = lane >> 4, lo = lane & 15;
  int m0 = blockIdx.x * 16, n0 = blockIdx.y * 16;
  CFrag acc;
#pragma unroll
  for (int d = 0; d < 8; ++d) acc.f[d] = 0.f;
  const float* wrow = W + (m0 + lo) * Cin;
  for (int k0 = 0; k0 < Cin; k0 += 32) {
    AFrag a, b;
#pragma unroll
    for (int e = 0; e < 16; ++e) a.h[e] = (_Float16)wrow[k0 + a_kidx(hi, e)];
#pragma unroll
    for (int e = 0; e < 16; ++e) b.h[e] = (_Float16)in[(k0 + b_kidx(hi, e)) * NPIX + n0 + lo];
    acc.v = __builtin_amdgcn_wmma_f32_16x16x32_f16(false, a.v, false, b.v,
                                                   (short)0, acc.v, false, false);
  }
  int n = n0 + lo;
#pragma unroll
  for (int d = 0; d < 8; ++d) {
    int m = m0 + d + (hi ? 8 : 0);
    float v = acc.f[d] + bias[m];
    if (HAS_RES) v += res[m * NPIX + n];
    out[m * NPIX + n] = fmaxf(v, 0.f);
  }
}

// ---------------------------------------------------------------------------
// bucket-conditioned local conv as per-bucket GEMM.
// grid = (72 buckets, 4 m-tiles), 32 threads (one wave) per block.
// A = emb[bucket] tile [16 x 576] staged to LDS by the TDM (f32, stride-577
// 2D tile -> packed rows), overlapped with the im2col gather of B.
// B = patches [576 x 16 pixels of this bucket] (LDS, f16, reflect pad).
// K = 576 = 18 WMMA steps.  Writes relu(localconv + bias).
// ---------------------------------------------------------------------------
__global__ void k_localconv_wmma(const float* __restrict__ emb, const float* __restrict__ gin,
                                 const int* __restrict__ counts, const int* __restrict__ offsets,
                                 const int* __restrict__ pixlist, float* __restrict__ out) {
  __shared__ float    AshF[16 * KW];   // f32 tile written by TDM (or fallback)
  __shared__ _Float16 Bsh[KW * 16];
  int lane = threadIdx.x;
  int hi = lane >> 4, lo = lane & 15;
  int b = blockIdx.x;
  int m0 = blockIdx.y * 16;
  const float* embrow = emb + b * EMBD;
  const float* gA = embrow + (size_t)m0 * KLOC;   // tile start (row m0, k 0)
  __builtin_prefetch(gA, 0, 3);                   // global_prefetch_b8

#if HAVE_TDM
  // --- Tensor DMA: 2D tile [16 rows x 576], element f32, row stride 577 ---
  {
    unsigned long long ga = (unsigned long long)(uintptr_t)gA;
    unsigned lds = (unsigned)(uintptr_t)(void*)AshF;   // LDS byte offset (addr[31:0])
    u32x4 g0;
    g0[0] = 1u;                                        // count=1, user descriptor
    g0[1] = lds;                                       // D#.lds_addr
    g0[2] = (unsigned)(ga & 0xffffffffu);              // D#.global_addr[31:0]
    g0[3] = (unsigned)((ga >> 32) & 0x01ffffffu) |     // D#.global_addr[56:32]
            (2u << 30);                                // D#.type = 2 ("image")
    i32x8 g1;
    g1[0] = (int)(2u << 16);                           // data_size=4B, no multicast
    g1[1] = (int)(((unsigned)KLOC & 0xffffu) << 16);   // tensor_dim0[15:0]
    g1[2] = (int)(((unsigned)KLOC >> 16) |             // tensor_dim0[31:16]
                  (16u << 16));                        // tensor_dim1[15:0] = 16 rows
    g1[3] = (int)((unsigned)KW << 16);                 // tensor_dim1[31:16]=0, tile_dim0=576
    g1[4] = (int)16;                                   // tile_dim1=16, tile_dim2=0
    g1[5] = (int)KLOC;                                 // tensor_dim0_stride[31:0] = 577
    g1[6] = 0;                                         // stride0 hi, stride1 lo
    g1[7] = 0;                                         // stride1 hi
    i32x4 gz  = {0, 0, 0, 0};                          // groups 2/3 unused (2D tensor)
    i32x8 gz8 = {0, 0, 0, 0, 0, 0, 0, 0};              // extra group (6-arg toolchain)
    __builtin_amdgcn_tensor_load_to_lds(g0, g1, gz, gz, gz8, 0);
  }
#else
  for (int idx = lane; idx < 16 * KW; idx += 32) {
    int r = idx / KW, k = idx - r * KW;
    AshF[idx] = embrow[(m0 + r) * KLOC + k];
  }
#endif

  int cnt = counts[b];
  int off = offsets[b];
  __syncthreads();

  for (int nb = 0; nb * 16 < cnt; ++nb) {
    // stage B tile: im2col with reflect padding; pad missing columns with 0.
    // This VALU/VMEM gather overlaps the in-flight TDM transfer of A.
    for (int idx = lane; idx < KW * 16; idx += 32) {
      int k = idx >> 4, col = idx & 15;
      float val = 0.f;
      int nc = nb * 16 + col;
      if (nc < cnt) {
        int p = pixlist[off + nc];
        int c = k / 9, r = k - 9 * c;
        int i = r / 3, j = r - 3 * i;
        int y = p / HWD, x = p - y * HWD;
        int yy = reflect48(y + i - 1), xx = reflect48(x + j - 1);
        val = gin[c * NPIX + yy * HWD + xx];
      }
      Bsh[idx] = (_Float16)val;
    }
#if HAVE_TDM
    __builtin_amdgcn_s_wait_tensorcnt((short)0);   // A tile landed in LDS
#endif
    __syncthreads();

    CFrag acc;
#pragma unroll
    for (int d = 0; d < 8; ++d) acc.f[d] = 0.f;
    for (int ks = 0; ks < KW / 32; ++ks) {   // 18 WMMA steps
      AFrag a, bf;
#pragma unroll
      for (int e = 0; e < 16; ++e)
        a.h[e] = (_Float16)AshF[lo * KW + ks * 32 + a_kidx(hi, e)];
#pragma unroll
      for (int e = 0; e < 16; ++e)
        bf.h[e] = Bsh[(ks * 32 + b_kidx(hi, e)) * 16 + lo];
      acc.v = __builtin_amdgcn_wmma_f32_16x16x32_f16(false, a.v, false, bf.v,
                                                     (short)0, acc.v, false, false);
    }
    int nc = nb * 16 + lo;
    if (nc < cnt) {
      int p = pixlist[off + nc];
#pragma unroll
      for (int d = 0; d < 8; ++d) {
        int m = m0 + d + (hi ? 8 : 0);
        float v = acc.f[d] + embrow[m * KLOC + KW];  // bias column
        out[m * NPIX + p] = fmaxf(v, 0.f);           // relu before body2 1x1
      }
    }
    __syncthreads();
  }
}

// ---------------------------------------------------------------------------
// host orchestration
// ---------------------------------------------------------------------------
struct EresP { const float *gw, *gb, *emb, *pw, *pb; };

static inline void run_conv1x1(hipStream_t s, const float* W, const float* b,
                               const float* in, const float* res, float* out, int Cin) {
  if (res)
    k_conv1x1_wmma<true><<<dim3(4, NPIX / 16), 32, 0, s>>>(W, b, in, res, out, Cin);
  else
    k_conv1x1_wmma<false><<<dim3(4, NPIX / 16), 32, 0, s>>>(W, b, in, nullptr, out, Cin);
}

static void run_eres(hipStream_t s, const EresP& p, const float* in, float* out,
                     float* gout, float* lout,
                     const int* counts, const int* offsets, const int* pixlist) {
  k_conv3x3<<<(64 * NPIX + 255) / 256, 256, 0, s>>>(in, p.gw, p.gb, gout, 64, 64, 4, 1);
  k_localconv_wmma<<<dim3(NTB, 4), 32, 0, s>>>(p.emb, gout, counts, offsets, pixlist, lout);
  run_conv1x1(s, p.pw, p.pb, lout, in, out, 64);  // relu(1x1 + b + x)
}

static void run_block(hipStream_t s, const EresP* ep,
                      const float* const* cw, const float* const* cb,
                      const float* in, float* out, float* blkcat,
                      float* o1, float* o2, float* gout, float* lout,
                      const int* counts, const int* offsets, const int* pixlist) {
  // blkcat channel layout: [c0 | b1 | b2 | b3] -> concat is free
  k_copy<<<(64 * NPIX + 255) / 256, 256, 0, s>>>(blkcat, in, 64 * NPIX);
  run_eres(s, ep[0], blkcat, blkcat + 64 * NPIX, gout, lout, counts, offsets, pixlist);
  run_conv1x1(s, cw[0], cb[0], blkcat, nullptr, o1, 128);
  run_eres(s, ep[1], o1, blkcat + 128 * NPIX, gout, lout, counts, offsets, pixlist);
  run_conv1x1(s, cw[1], cb[1], blkcat, nullptr, o2, 192);
  run_eres(s, ep[2], o2, blkcat + 192 * NPIX, gout, lout, counts, offsets, pixlist);
  run_conv1x1(s, cw[2], cb[2], blkcat, nullptr, out, 256);
}

extern "C" void kernel_launch(void* const* d_in, const int* in_sizes, int n_in,
                              void* d_out, int out_size, void* d_ws, size_t ws_size,
                              hipStream_t stream) {
  (void)in_sizes; (void)n_in; (void)out_size; (void)ws_size;
  auto F = [&](int i) { return (const float*)d_in[i]; };

  const float* x       = F(0);
  const int*   buckets = (const int*)d_in[1];
  const float* entry_w = F(2);
  const float* entry_b = F(3);

  EresP eres[3][3];
  const float* cw[3][3];
  const float* cb[3][3];
  for (int bi = 0; bi < 3; ++bi) {
    int base = 4 + bi * 21;
    for (int e = 0; e < 3; ++e) {
      int eb = base + e * 5;
      eres[bi][e] = { F(eb), F(eb + 1), F(eb + 2), F(eb + 3), F(eb + 4) };
    }
    for (int j = 0; j < 3; ++j) {
      cw[bi][j] = F(base + 15 + j * 2);
      cb[bi][j] = F(base + 15 + j * 2 + 1);
    }
  }
  const float* ncw[3]; const float* ncb[3];
  for (int j = 0; j < 3; ++j) { ncw[j] = F(67 + j * 2); ncb[j] = F(67 + j * 2 + 1); }
  const float* exit_w = F(73);
  const float* exit_b = F(74);

  // workspace layout (floats)
  float* fws = (float*)d_ws;
  size_t off = 0;
  float* netcat = fws + off; off += (size_t)256 * NPIX;   // [c0|B1|B2|B3]
  float* blkcat = fws + off; off += (size_t)256 * NPIX;   // per-block concat
  float* o1n = fws + off; off += (size_t)64 * NPIX;
  float* o2n = fws + off; off += (size_t)64 * NPIX;
  float* o1b = fws + off; off += (size_t)64 * NPIX;
  float* o2b = fws + off; off += (size_t)64 * NPIX;
  float* gout = fws + off; off += (size_t)64 * NPIX;
  float* lout = fws + off; off += (size_t)64 * NPIX;
  int* iws = (int*)(fws + off);
  int* counts  = iws;
  int* cursor  = iws + 128;
  int* offsets = iws + 256;       // NTB+1
  int* pixlist = iws + 384;       // NPIX

  // bucket binning (buckets are constant across all 9 local-conv layers)
  k_zero_bins<<<1, 128, 0, stream>>>(counts, cursor, offsets);
  k_hist<<<(NPIX + 255) / 256, 256, 0, stream>>>(buckets, counts);
  k_scan<<<1, 32, 0, stream>>>(counts, offsets);
  k_fill<<<(NPIX + 255) / 256, 256, 0, stream>>>(buckets, offsets, cursor, pixlist);

  // entry conv -> netcat slot 0 (c0)
  k_conv3x3<<<(64 * NPIX + 255) / 256, 256, 0, stream>>>(x, entry_w, entry_b,
                                                         netcat, 3, 64, 1, 0);

  run_block(stream, eres[0], cw[0], cb[0], netcat, netcat + 64 * NPIX,
            blkcat, o1b, o2b, gout, lout, counts, offsets, pixlist);
  run_conv1x1(stream, ncw[0], ncb[0], netcat, nullptr, o1n, 128);

  run_block(stream, eres[1], cw[1], cb[1], o1n, netcat + 128 * NPIX,
            blkcat, o1b, o2b, gout, lout, counts, offsets, pixlist);
  run_conv1x1(stream, ncw[1], ncb[1], netcat, nullptr, o2n, 192);

  run_block(stream, eres[2], cw[2], cb[2], o2n, netcat + 192 * NPIX,
            blkcat, o1b, o2b, gout, lout, counts, offsets, pixlist);
  run_conv1x1(stream, ncw[2], ncb[2], netcat, nullptr, o1n, 256);  // o3 reuses o1n

  // exit conv -> d_out [3, 48, 48]
  k_conv3x3<<<(3 * NPIX + 255) / 256, 256, 0, stream>>>(o1n, exit_w, exit_b,
                                                        (float*)d_out, 64, 3, 1, 0);
}
